// TacotronSTFT_79748952752369
// MI455X (gfx1250) — matrix operand order repaired and checked by reading, not compiled, not run on hardware
//
#include <hip/hip_runtime.h>
#include <math.h>

typedef __attribute__((ext_vector_type(16))) _Float16 v16h;
typedef __attribute__((ext_vector_type(8)))  _Float16 v8h;
typedef __attribute__((ext_vector_type(8)))  float    v8f;

// generic-vector int4 matching the builtin's expected pointee type
typedef int v4i_g __attribute__((vector_size(16)));
typedef __attribute__((address_space(1))) v4i_g* as1_v4i;   // global
typedef __attribute__((address_space(3))) v4i_g* as3_v4i;   // LDS

#define KLEN      1024          // FILTER_LENGTH (GEMM K)
#define HOP       256
#define CUTOFF    513
#define NCH       1026          // 2*CUTOFF output channels
#define NCH_PAD   1056          // padded to 66 tiles of 16 (even pairs per wave)
#define NTILES    66
#define N_MELS    80
#define BATCH     32
#define NSAMP     220500
#define T_FRAMES  862           // (220500/256)+1
#define M_TOTAL   (BATCH * T_FRAMES)   // 27584, divisible by 16
#define MAG_STRIDE 520

__device__ __forceinline__ v16h cat16(v8h lo, v8h hi) {
    return __builtin_shufflevector(lo, hi, 0,1,2,3,4,5,6,7,
                                           8,9,10,11,12,13,14,15);
}

// ---------------------------------------------------------------------------
// Pack the windowed DFT basis (1026 x 1024 f32) into f16, padded to 1056 rows.
// ---------------------------------------------------------------------------
__global__ void pack_basis_kernel(const float* __restrict__ fb,
                                  _Float16* __restrict__ Bmat) {
    int idx = blockIdx.x * 256 + threadIdx.x;
    if (idx >= NCH_PAD * KLEN) return;
    int c = idx >> 10;
    int k = idx & (KLEN - 1);
    float v = (c < NCH) ? fb[c * KLEN + k] : 0.0f;
    Bmat[idx] = (_Float16)v;
}

// ---------------------------------------------------------------------------
// Materialize frame matrix A[27584][1024] in f16, applying reflect padding.
// Row r = b*862 + t ; A[r][k] = y[b][ reflect(t*256 + k - 512) ]
// ---------------------------------------------------------------------------
__global__ void pack_frames_kernel(const float* __restrict__ y,
                                   _Float16* __restrict__ Amat) {
    size_t idx = (size_t)blockIdx.x * 256 + threadIdx.x;
    if (idx >= (size_t)M_TOTAL * KLEN) return;
    int r = (int)(idx >> 10);
    int k = (int)(idx & (KLEN - 1));
    int b = r / T_FRAMES;
    int t = r - b * T_FRAMES;
    int s = t * HOP + k - (KLEN / 2);
    if (s < 0) s = -s;                               // reflect left
    if (s >= NSAMP) s = 2 * NSAMP - 2 - s;           // reflect right
    Amat[idx] = (_Float16)y[(size_t)b * NSAMP + s];
}

// ---------------------------------------------------------------------------
// Fused STFT-GEMM + magnitude + mel + log + energy.
// One workgroup (256 threads = 8 waves) owns 16 consecutive frames.
// Each wave processes N-tile PAIRS: one A fragment feeds two WMMA chains;
// fragment loads are double-buffered so loads overlap the matrix ops.
// ---------------------------------------------------------------------------
__global__ __launch_bounds__(256) void stft_mel_kernel(
    const _Float16* __restrict__ Amat,
    const _Float16* __restrict__ Bmat,
    const float*    __restrict__ melb,
    float*          __restrict__ out) {

    extern __shared__ char smem[];
    _Float16* As = (_Float16*)smem;                                  // 16*1024 f16
    float*    ft = (float*)(smem + 16 * KLEN * sizeof(_Float16));    // 16*1056 f32
    float*    mg = (float*)(smem + 16 * KLEN * sizeof(_Float16)
                                 + 16 * NCH_PAD * sizeof(float));    // 16*520 f32

    const int tid  = threadIdx.x;
    const int lane = tid & 31;
    const int wave = tid >> 5;
    const int row0 = blockIdx.x * 16;

    // ---- stage A tile into LDS: async VGPR-bypassing copy when available ----
    {
        const uint4* src = (const uint4*)(Amat + (size_t)row0 * KLEN);
        uint4*       dst = (uint4*)As;
#if __has_builtin(__builtin_amdgcn_global_load_async_to_lds_b128)
        for (int i = tid; i < (16 * KLEN * 2) / 16; i += 256) {
            __builtin_amdgcn_global_load_async_to_lds_b128(
                (as1_v4i)(uintptr_t)(src + i),
                (as3_v4i)(unsigned int)(uintptr_t)(dst + i),
                /*offset=*/0, /*cpol=*/0);
        }
        asm volatile("s_wait_asynccnt 0x0" ::: "memory");
#else
        #pragma unroll 2
        for (int i = tid; i < (16 * KLEN * 2) / 16; i += 256) dst[i] = src[i];
#endif
    }
    __syncthreads();

    const int mrow = lane & 15;       // A row / B column within tile
    const int hsel = lane >> 4;       // which K half-group this lane holds

    // 16-bit A layout: lane holds K chunks {hsel*8..+7} and {hsel*8+16..+23}
    const _Float16* Abase = As + mrow * KLEN + hsel * 8;

    // wave w owns tile pairs {2w, 2w+1} stepping by 16 -> all 66 tiles covered
    for (int nt = wave * 2; nt < NTILES; nt += 16) {
        const _Float16* Bb0 = Bmat + (size_t)(nt * 16 + mrow) * KLEN + hsel * 8;
        const _Float16* Bb1 = Bb0 + 16 * KLEN;   // adjacent N-tile, same lane col

        v8f acc0 = {};
        v8f acc1 = {};

        // -------- software-pipelined K loop (double-buffered fragments) ------
        v8h a0 = *(const v8h*)(Abase);
        v8h a1 = *(const v8h*)(Abase + 16);
        v8h b0 = *(const v8h*)(Bb0);
        v8h b1 = *(const v8h*)(Bb0 + 16);
        v8h c0 = *(const v8h*)(Bb1);
        v8h c1 = *(const v8h*)(Bb1 + 16);

        #pragma unroll 2
        for (int kk = 0; kk < KLEN - 32; kk += 32) {
            v8h na0 = *(const v8h*)(Abase + kk + 32);
            v8h na1 = *(const v8h*)(Abase + kk + 48);
            v8h nb0 = *(const v8h*)(Bb0 + kk + 32);
            v8h nb1 = *(const v8h*)(Bb0 + kk + 48);
            v8h nc0 = *(const v8h*)(Bb1 + kk + 32);
            v8h nc1 = *(const v8h*)(Bb1 + kk + 48);

            v16h af = cat16(a0, a1);
            acc0 = __builtin_amdgcn_wmma_f32_16x16x32_f16(
                       false, af, false, cat16(b0, b1), (short)0, acc0, false, false);
            acc1 = __builtin_amdgcn_wmma_f32_16x16x32_f16(
                       false, af, false, cat16(c0, c1), (short)0, acc1, false, false);

            a0 = na0; a1 = na1; b0 = nb0; b1 = nb1; c0 = nc0; c1 = nc1;
        }
        {   // final K step
            v16h af = cat16(a0, a1);
            acc0 = __builtin_amdgcn_wmma_f32_16x16x32_f16(
                       false, af, false, cat16(b0, b1), (short)0, acc0, false, false);
            acc1 = __builtin_amdgcn_wmma_f32_16x16x32_f16(
                       false, af, false, cat16(c0, c1), (short)0, acc1, false, false);
        }

        // C/D layout: VGPR j -> M = j + 8*hsel, N = lane&15
        const int ncol0 = nt * 16 + mrow;
        const int mbase = hsel * 8;
        #pragma unroll
        for (int j = 0; j < 8; ++j) {
            ft[(mbase + j) * NCH_PAD + ncol0]      = acc0[j];
            ft[(mbase + j) * NCH_PAD + ncol0 + 16] = acc1[j];
        }
    }
    __syncthreads();

    // ---- magnitude: sqrt(re^2 + im^2), channel c pairs with c+513 ----
    for (int i = tid; i < 16 * CUTOFF; i += 256) {
        int m = i / CUTOFF;
        int c = i - m * CUTOFF;
        float re = ft[m * NCH_PAD + c];
        float im = ft[m * NCH_PAD + c + CUTOFF];
        mg[m * MAG_STRIDE + c] = sqrtf(re * re + im * im);
    }
    __syncthreads();

    // ---- mel projection + log ----
    for (int i = tid; i < 16 * N_MELS; i += 256) {
        int m = i / N_MELS;
        int j = i - m * N_MELS;
        const float* mb = melb + j * CUTOFF;
        const float* mr = mg + m * MAG_STRIDE;
        float s = 0.0f;
        #pragma unroll 4
        for (int c = 0; c < CUTOFF; ++c) s = fmaf(mb[c], mr[c], s);
        int r = row0 + m;
        int b = r / T_FRAMES;
        int t = r - b * T_FRAMES;
        out[((size_t)b * N_MELS + j) * T_FRAMES + t] = logf(fmaxf(s, 1e-5f));
    }

    // ---- energy: L2 norm over frequency bins ----
    if (tid < 16) {
        int m = tid;
        const float* mr = mg + m * MAG_STRIDE;
        float s = 0.0f;
        #pragma unroll 4
        for (int c = 0; c < CUTOFF; ++c) s = fmaf(mr[c], mr[c], s);
        int r = row0 + m;
        int b = r / T_FRAMES;
        int t = r - b * T_FRAMES;
        out[(size_t)BATCH * N_MELS * T_FRAMES + (size_t)b * T_FRAMES + t] = sqrtf(s);
    }
}

// ---------------------------------------------------------------------------
extern "C" void kernel_launch(void* const* d_in, const int* in_sizes, int n_in,
                              void* d_out, int out_size, void* d_ws, size_t ws_size,
                              hipStream_t stream) {
    const float* y    = (const float*)d_in[0];   // (32, 220500) f32
    const float* fb   = (const float*)d_in[1];   // (1026, 1, 1024) f32
    const float* melb = (const float*)d_in[2];   // (80, 513) f32
    float*       out  = (float*)d_out;           // mel (32,80,862) ++ energy (32,862)

    _Float16* Amat = (_Float16*)d_ws;                          // 27584*1024 f16
    _Float16* Bmat = Amat + (size_t)M_TOTAL * KLEN;            // 1056*1024  f16

    {
        int n = M_TOTAL * KLEN;  // 28,246,016
        pack_frames_kernel<<<(n + 255) / 256, 256, 0, stream>>>(y, Amat);
    }
    {
        int n = NCH_PAD * KLEN;  // 1,081,344
        pack_basis_kernel<<<(n + 255) / 256, 256, 0, stream>>>(fb, Bmat);
    }
    {
        size_t smem = 16 * KLEN * sizeof(_Float16)     // 32768  A tile
                    + 16 * NCH_PAD * sizeof(float)     // 67584  ft
                    + 16 * MAG_STRIDE * sizeof(float); // 33280  mag
        stft_mel_kernel<<<M_TOTAL / 16, 256, smem, stream>>>(Amat, Bmat, melb, out);
    }
}
